// MultiMessagePassingWithAttention_4870492914027
// MI455X (gfx1250) — matrix operand order repaired and checked by reading, not compiled, not run on hardware
//
#include <hip/hip_runtime.h>
#include <hip/hip_bf16.h>

#define EMB   128
#define EDGE  32
#define STEPS 3
#define HEADS 3
#define Gg    64
#define Ll    48
#define Nn    (Gg * Ll)      // 3072 nodes
#define Ee    (Nn * 16)      // 49152 edges

typedef _Float16 f16;
typedef f16  v16h __attribute__((ext_vector_type(16)));
typedef f16  v8h  __attribute__((ext_vector_type(8)));
typedef float v8f __attribute__((ext_vector_type(8)));

__device__ __forceinline__ float leaky(float v, float s) { return v >= 0.f ? v : s * v; }

// ---- A-fragment loader: 16-bit A 16x32 layout ----
// lanes 0-15 hold K {0..7, 16..23}; lanes 16-31 hold K {8..15, 24..31}
// rowp points at this lane's row, at local K offset 0 of the 32-wide chunk.
__device__ __forceinline__ v16h load_a_frag(const f16* rowp, int khalf) {
    v8h lo = *(const v8h*)(rowp + khalf * 8);
    v8h hi = *(const v8h*)(rowp + 16 + khalf * 8);
    return __builtin_shufflevector(lo, hi, 0,1,2,3,4,5,6,7,8,9,10,11,12,13,14,15);
}

__device__ __forceinline__ v8f wmma_f16(v16h a, v16h b, v8f c) {
    return __builtin_amdgcn_wmma_f32_16x16x32_f16(false, a, false, b, (short)0, c, false, false);
}

// float -> order-preserving int key (involution)
__device__ __forceinline__ int fkey(float f) {
    int b = __float_as_int(f);
    return b >= 0 ? b : (b ^ 0x7FFFFFFF);
}

// ================= prep kernels =================
__global__ void cvt_init_kernel(const float* __restrict__ xin, float* __restrict__ xcur,
                                f16* __restrict__ xh, int n) {
    int t = blockIdx.x * blockDim.x + threadIdx.x;
    if (t < n) { float v = xin[t]; xcur[t] = v; xh[t] = (f16)v; }
}
__global__ void cvt_h_kernel(const float* __restrict__ a, f16* __restrict__ b, int n) {
    int t = blockIdx.x * blockDim.x + threadIdx.x;
    if (t < n) b[t] = (f16)a[t];
}
__global__ void zero_h_kernel(f16* __restrict__ a, int n) {
    int t = blockIdx.x * blockDim.x + threadIdx.x;
    if (t < n) a[t] = (f16)0.f;
}
__global__ void init_key_kernel(int* __restrict__ a, int n) {
    int t = blockIdx.x * blockDim.x + threadIdx.x;
    if (t < n) a[t] = INT_MIN;
}
__global__ void copy_out_kernel(const float* __restrict__ a, float* __restrict__ b, int n) {
    int t = blockIdx.x * blockDim.x + threadIdx.x;
    if (t < n) b[t] = a[t];
}

// Swizzle fp32 [K x Nc] row-major weights into f16 WMMA B-fragments.
// B layout (16-bit, 32x16): lane l holds column n = l%16; element e -> K = (e&7) + 8*(l>=16) + 16*(e>=8).
__global__ void swz_kernel(const float* __restrict__ W, f16* __restrict__ out, int K, int Nc) {
    int t = blockIdx.x * blockDim.x + threadIdx.x;
    int total = K * Nc;
    if (t >= total) return;
    int e    = t & 15;
    int l    = (t >> 4) & 31;
    int rest = t >> 9;
    int Nt = Nc >> 4;
    int w  = rest % Nt;
    int kc = rest / Nt;
    int k = kc * 32 + ((e & 8) ? 16 : 0) + ((l >= 16) ? 8 : 0) + (e & 7);
    int n = w * 16 + (l & 15);
    out[t] = (f16)W[(size_t)k * Nc + n];
}

// ============ message GEMM + fused scatter-max ============
// msg = leaky([x[src]|edge_attr] @ Wm + bm);  agg[dst] = max(agg[dst], msg)
// grid.x = E/16 (M-tiles), 256 threads = 8 waves; wave w -> N-tile w (128 cols / 16)
__global__ void msg_kernel(const f16* __restrict__ xh, const f16* __restrict__ eh,
                           const f16* __restrict__ Wswz, const float* __restrict__ bias,
                           const int* __restrict__ src, const int* __restrict__ dst,
                           int* __restrict__ aggk) {
    int lane = threadIdx.x & 31, wave = threadIdx.x >> 5;
    int etile = blockIdx.x;
    int r = lane & 15, khalf = lane >> 4;
    int e = etile * 16 + r;
    int s = src[e];
    const f16* xrow = xh + (size_t)s * EMB;
    const f16* erow = eh + (size_t)e * EDGE;
    v8f acc = {};
#pragma unroll
    for (int kc = 0; kc < 5; ++kc) {
        const f16* rowp = (kc < 4) ? (xrow + kc * 32) : erow;
        v16h a = load_a_frag(rowp, khalf);
        v16h b = *(const v16h*)(Wswz + (((size_t)kc * 8 + wave) * 32 + lane) * 16);
        acc = wmma_f16(a, b, acc);
    }
    int n = wave * 16 + (lane & 15);
    float bn = bias[n];
#pragma unroll
    for (int v = 0; v < 8; ++v) {
        int row = v + khalf * 8;
        float m = leaky(acc[v] + bn, 0.01f);
        int d = dst[etile * 16 + row];
        atomicMax(&aggk[(size_t)d * EMB + n], fkey(m));
    }
}

__global__ void aggfin_kernel(const int* __restrict__ key, f16* __restrict__ aggh, int n) {
    int t = blockIdx.x * blockDim.x + threadIdx.x;
    if (t >= n) return;
    int k = key[t];
    float v = 0.f;
    if (k != INT_MIN) { int b = k >= 0 ? k : (k ^ 0x7FFFFFFF); v = __int_as_float(b); }
    aggh[t] = (f16)v;
}

// ============ node update GEMM ============
// x = leaky([x|xg|agg] @ Wa + ba) + x ; grid.x = N/16, 256 threads, 8 waves = 8 N-tiles
__global__ void update_kernel(const f16* __restrict__ xh, const f16* __restrict__ xatth,
                              const f16* __restrict__ aggh, const f16* __restrict__ Wswz,
                              const float* __restrict__ bias, float* __restrict__ x) {
    int lane = threadIdx.x & 31, wave = threadIdx.x >> 5;
    int tile = blockIdx.x;
    int r = lane & 15, khalf = lane >> 4;
    int node = tile * 16 + r;
    int g = node / Ll;
    const f16* xrow = xh    + (size_t)node * EMB;
    const f16* grow = xatth + (size_t)g    * EMB;
    const f16* arow = aggh  + (size_t)node * EMB;
    v8f acc = {};
#pragma unroll
    for (int kc = 0; kc < 12; ++kc) {
        const f16* rowp;
        if (kc < 4)      rowp = xrow + kc * 32;
        else if (kc < 8) rowp = grow + (kc - 4) * 32;
        else             rowp = arow + (kc - 8) * 32;
        v16h a = load_a_frag(rowp, khalf);
        v16h b = *(const v16h*)(Wswz + (((size_t)kc * 8 + wave) * 32 + lane) * 16);
        acc = wmma_f16(a, b, acc);
    }
    int n = wave * 16 + (lane & 15);
    float bn = bias[n];
#pragma unroll
    for (int v = 0; v < 8; ++v) {
        int nd = tile * 16 + v + khalf * 8;
        size_t idx = (size_t)nd * EMB + n;
        x[idx] = leaky(acc[v] + bn, 0.01f) + x[idx];
    }
}

// ============ GATv2 projections: xl = x@Wl, xr = x@Wr ============
// K=128, Ncols=384 (24 N-tiles). grid=(N/16, 2); 8 waves x 3 tiles each.
__global__ void xlr_kernel(const f16* __restrict__ xh, const f16* __restrict__ WlS,
                           const f16* __restrict__ WrS, float* __restrict__ xl,
                           float* __restrict__ xr) {
    const f16* Wswz = blockIdx.y ? WrS : WlS;
    float* out = blockIdx.y ? xr : xl;
    int lane = threadIdx.x & 31, wave = threadIdx.x >> 5;
    int tile = blockIdx.x;
    int r = lane & 15, khalf = lane >> 4;
    int node = tile * 16 + r;
    const f16* xrow = xh + (size_t)node * EMB;
    v16h a0 = load_a_frag(xrow +  0, khalf);
    v16h a1 = load_a_frag(xrow + 32, khalf);
    v16h a2 = load_a_frag(xrow + 64, khalf);
    v16h a3 = load_a_frag(xrow + 96, khalf);
#pragma unroll
    for (int t = 0; t < 3; ++t) {
        int w = wave + t * 8;   // N-tile 0..23
        v8f acc = {};
        acc = wmma_f16(a0, *(const v16h*)(WlS == Wswz || 1 ? Wswz + (((size_t)0 * 24 + w) * 32 + lane) * 16 : 0), acc);
        acc = wmma_f16(a1, *(const v16h*)(Wswz + (((size_t)1 * 24 + w) * 32 + lane) * 16), acc);
        acc = wmma_f16(a2, *(const v16h*)(Wswz + (((size_t)2 * 24 + w) * 32 + lane) * 16), acc);
        acc = wmma_f16(a3, *(const v16h*)(Wswz + (((size_t)3 * 24 + w) * 32 + lane) * 16), acc);
        int n = w * 16 + (lane & 15);
#pragma unroll
        for (int v = 0; v < 8; ++v)
            out[(size_t)(tile * 16 + v + khalf * 8) * 384 + n] = acc[v];
    }
}

// ============ GATv2 edge attention + softmax + aggregate ============
// 1 wave per dst node, 4 waves per block; complete graph of 48 src within block.
__global__ void attn_kernel(const float* __restrict__ xl, const float* __restrict__ xr,
                            const float* __restrict__ att_a, const float* __restrict__ att_b,
                            f16* __restrict__ xatth) {
    __shared__ float eLDS[4][HEADS * Ll];
    int lane = threadIdx.x & 31, wave = threadIdx.x >> 5;
    int d = blockIdx.x * 4 + wave;
    int g = d / Ll;
    int base = g * Ll;
    int c0 = lane * 4;
    float xrd[HEADS][4], av[HEADS][4];
#pragma unroll
    for (int h = 0; h < HEADS; ++h)
#pragma unroll
        for (int k = 0; k < 4; ++k) {
            xrd[h][k] = xr[(size_t)d * 384 + h * EMB + c0 + k];
            av[h][k]  = att_a[h * EMB + c0 + k];
        }
    float m[HEADS] = {-3.0e38f, -3.0e38f, -3.0e38f};
    for (int s = 0; s < Ll; ++s) {
        const float* xls = xl + (size_t)(base + s) * 384;
#pragma unroll
        for (int h = 0; h < HEADS; ++h) {
            float part = 0.f;
#pragma unroll
            for (int k = 0; k < 4; ++k) {
                float z = xls[h * EMB + c0 + k] + xrd[h][k];
                z = z >= 0.f ? z : 0.2f * z;
                part += z * av[h][k];
            }
            for (int off = 16; off > 0; off >>= 1) part += __shfl_xor(part, off);
            if (lane == 0) eLDS[wave][s * HEADS + h] = part;
            m[h] = fmaxf(m[h], part);
        }
    }
    __syncthreads();
    float sum[HEADS] = {0.f, 0.f, 0.f};
    float outc[HEADS][4] = {};
    for (int s = 0; s < Ll; ++s) {
        const float* xls = xl + (size_t)(base + s) * 384;
#pragma unroll
        for (int h = 0; h < HEADS; ++h) {
            float p = __expf(eLDS[wave][s * HEADS + h] - m[h]);
            sum[h] += p;
#pragma unroll
            for (int k = 0; k < 4; ++k) outc[h][k] += p * xls[h * EMB + c0 + k];
        }
    }
#pragma unroll
    for (int k = 0; k < 4; ++k) {
        float o = (outc[0][k] / sum[0] + outc[1][k] / sum[1] + outc[2][k] / sum[2]) * (1.f / 3.f)
                  + att_b[c0 + k];
        o = leaky(o, 0.01f);
        xatth[(size_t)d * EMB + c0 + k] = (f16)o;
    }
}

// ============ global attention pooling ============
// 1 block per graph, 128 threads (4 waves).
__global__ void pool_kernel(const float* __restrict__ x,
                            const float* __restrict__ Wg, const float* __restrict__ bg,
                            const float* __restrict__ Wf, const float* __restrict__ bfv,
                            const float* __restrict__ Wt, const float* __restrict__ bt,
                            float* __restrict__ out) {
    __shared__ float alpha[Ll];
    __shared__ float xg[EMB];
    __shared__ float ssum;
    int g = blockIdx.x;
    int tid = threadIdx.x;
    int lane = tid & 31, wave = tid >> 5;
    // gates
    for (int j = wave; j < Ll; j += 4) {
        const float* xrow = x + (size_t)(g * Ll + j) * EMB;
        float part = 0.f;
#pragma unroll
        for (int k = 0; k < 4; ++k) part += xrow[lane * 4 + k] * Wg[lane * 4 + k];
        for (int off = 16; off > 0; off >>= 1) part += __shfl_xor(part, off);
        if (lane == 0) alpha[j] = part + bg[0];
    }
    __syncthreads();
    if (tid == 0) {
        float mx = -3.0e38f;
        for (int j = 0; j < Ll; ++j) mx = fmaxf(mx, alpha[j]);
        float s = 0.f;
        for (int j = 0; j < Ll; ++j) { float p = __expf(alpha[j] - mx); alpha[j] = p; s += p; }
        ssum = s;
    }
    __syncthreads();
    {   // feat matvecs + weighted sum
        int c = tid;
        float inv = 1.f / ssum;
        float acc = 0.f;
        for (int j = 0; j < Ll; ++j) {
            const float* xrow = x + (size_t)(g * Ll + j) * EMB;
            float f = bfv[c];
            for (int k = 0; k < EMB; ++k) f += xrow[k] * Wf[(size_t)k * EMB + c];
            acc += alpha[j] * inv * leaky(f, 0.01f);
        }
        xg[c] = acc;
    }
    __syncthreads();
    {
        int c = tid;
        float t = bt[c];
        for (int k = 0; k < EMB; ++k) t += xg[k] * Wt[(size_t)k * EMB + c];
        out[(size_t)g * EMB + c] = leaky(t, 0.01f);
    }
}

// ================= host launcher =================
static inline size_t alignup(size_t v) { return (v + 255) & ~(size_t)255; }

extern "C" void kernel_launch(void* const* d_in, const int* in_sizes, int n_in,
                              void* d_out, int out_size, void* d_ws, size_t ws_size,
                              hipStream_t stream) {
    const float* x_in  = (const float*)d_in[0];
    const float* ea_in = (const float*)d_in[1];
    const int*   eidx  = (const int*)d_in[2];
    const float* Wm = (const float*)d_in[6];
    const float* bm = (const float*)d_in[7];
    const float* Wa = (const float*)d_in[8];
    const float* ba = (const float*)d_in[9];
    const float* Wl = (const float*)d_in[10];
    const float* Wr = (const float*)d_in[11];
    const float* aa = (const float*)d_in[12];
    const float* ab = (const float*)d_in[13];
    const float* Wg = (const float*)d_in[14];
    const float* bg = (const float*)d_in[15];
    const float* Wf = (const float*)d_in[16];
    const float* bf = (const float*)d_in[17];
    const float* Wt = (const float*)d_in[18];
    const float* bt = (const float*)d_in[19];
    float* out = (float*)d_out;
    (void)in_sizes; (void)n_in; (void)out_size; (void)ws_size;

    char* p = (char*)d_ws; size_t off = 0;
    auto carve = [&](size_t bytes) -> char* { char* r = p + off; off = alignup(off + bytes); return r; };
    float* x_cur = (float*)carve((size_t)Nn * EMB * 4);
    f16*   xh    = (f16*)  carve((size_t)Nn * EMB * 2);
    f16*   eh    = (f16*)  carve((size_t)Ee * EDGE * 2);
    int*   aggk  = (int*)  carve((size_t)Nn * EMB * 4);
    f16*   aggh  = (f16*)  carve((size_t)Nn * EMB * 2);
    f16*   xatth = (f16*)  carve((size_t)Nn * EMB * 2);
    float* xl    = (float*)carve((size_t)Nn * 384 * 4);
    float* xr    = (float*)carve((size_t)Nn * 384 * 4);
    f16*   WmS   = (f16*)  carve((size_t)3 * 160 * 128 * 2);
    f16*   WaS   = (f16*)  carve((size_t)3 * 384 * 128 * 2);
    f16*   WlS   = (f16*)  carve((size_t)2 * 128 * 384 * 2);
    f16*   WrS   = (f16*)  carve((size_t)2 * 128 * 384 * 2);

    const int* srcp = eidx;
    const int* dstp = eidx + Ee;

    cvt_init_kernel<<<(Nn * EMB + 255) / 256, 256, 0, stream>>>(x_in, x_cur, xh, Nn * EMB);
    zero_h_kernel<<<(Nn * EMB + 255) / 256, 256, 0, stream>>>(xatth, Nn * EMB);
    cvt_h_kernel<<<(Ee * EDGE + 255) / 256, 256, 0, stream>>>(ea_in, eh, Ee * EDGE);
    for (int i = 0; i < 3; ++i) {
        swz_kernel<<<(160 * 128 + 255) / 256, 256, 0, stream>>>(Wm + (size_t)i * 160 * 128, WmS + (size_t)i * 160 * 128, 160, 128);
        swz_kernel<<<(384 * 128 + 255) / 256, 256, 0, stream>>>(Wa + (size_t)i * 384 * 128, WaS + (size_t)i * 384 * 128, 384, 128);
    }
    for (int i = 0; i < 2; ++i) {
        swz_kernel<<<(128 * 384 + 255) / 256, 256, 0, stream>>>(Wl + (size_t)i * 128 * 384, WlS + (size_t)i * 128 * 384, 128, 384);
        swz_kernel<<<(128 * 384 + 255) / 256, 256, 0, stream>>>(Wr + (size_t)i * 128 * 384, WrS + (size_t)i * 128 * 384, 128, 384);
    }

    for (int i = 0; i < STEPS; ++i) {
        init_key_kernel<<<(Nn * EMB + 255) / 256, 256, 0, stream>>>(aggk, Nn * EMB);
        msg_kernel<<<Ee / 16, 256, 0, stream>>>(xh, eh, WmS + (size_t)i * 160 * 128,
                                                bm + i * EMB, srcp, dstp, aggk);
        aggfin_kernel<<<(Nn * EMB + 255) / 256, 256, 0, stream>>>(aggk, aggh, Nn * EMB);
        update_kernel<<<Nn / 16, 256, 0, stream>>>(xh, xatth, aggh,
                                                   WaS + (size_t)i * 384 * 128, ba + i * EMB, x_cur);
        cvt_h_kernel<<<(Nn * EMB + 255) / 256, 256, 0, stream>>>(x_cur, xh, Nn * EMB);
        if (i < STEPS - 1) {
            dim3 grid(Nn / 16, 2);
            xlr_kernel<<<grid, 256, 0, stream>>>(xh, WlS + (size_t)i * 128 * 384,
                                                 WrS + (size_t)i * 128 * 384, xl, xr);
            attn_kernel<<<Nn / 4, 128, 0, stream>>>(xl, xr, aa + (size_t)i * HEADS * EMB,
                                                    ab + (size_t)i * EMB, xatth);
        }
    }
    copy_out_kernel<<<(Nn * EMB + 255) / 256, 256, 0, stream>>>(x_cur, out, Nn * EMB);
    pool_kernel<<<Gg, 128, 0, stream>>>(x_cur, Wg, bg, Wf, bf, Wt, bt, out + (size_t)Nn * EMB);
}
